// KTMutualAttention_34239479284420
// MI455X (gfx1250) — matrix operand order repaired and checked by reading, not compiled, not run on hardware
//
#include <hip/hip_runtime.h>

#define DEV __device__ __forceinline__

typedef __attribute__((ext_vector_type(16))) __bf16 v16bf;
typedef __attribute__((ext_vector_type(8)))  __bf16 bf16x8;
typedef __attribute__((ext_vector_type(4)))  __bf16 bf16x4;
typedef __attribute__((ext_vector_type(8)))  float  v8f;
typedef __attribute__((ext_vector_type(4)))  float  f32x4;

static constexpr int BB = 8, T = 512, S = 1024, TL = 64, D = 1024, H = 16, HD = 64;
static constexpr float QSCALE = 0.125f;  // HEAD_DIM ** -0.5

DEV v8f wmma_bf16(v16bf a, v16bf b, v8f c) {
  return __builtin_amdgcn_wmma_f32_16x16x32_bf16(false, a, false, b, (short)0, c,
                                                 false, false);
}

DEV v8f vzero8() {
  v8f z;
#pragma unroll
  for (int r = 0; r < 8; ++r) z[r] = 0.0f;
  return z;
}

// A fragment 16x32 bf16 (row-major source). Caller passes per-lane pointer:
// base + (row m = lane&15)*lda + (lane>>4)*16.  Elements 0..7 at +0, 8..15 at +8.
DEV v16bf ld_frag_a(const __bf16* p) {
  bf16x8 lo = *(const bf16x8*)p;
  bf16x8 hi = *(const bf16x8*)(p + 8);
  v16bf r;
#pragma unroll
  for (int i = 0; i < 8; ++i) { r[i] = lo[i]; r[8 + i] = hi[i]; }
  return r;
}

// ---- gfx1250 async global->LDS copy (ASYNCcnt-tracked), 16B per lane --------
DEV void async_ld16(unsigned int lds_off, const void* gaddr) {
  asm volatile("global_load_async_to_lds_b128 %0, %1, off"
               :: "v"(lds_off), "v"(gaddr)
               : "memory");
}

#if defined(__AMDGCN__) && __has_builtin(__builtin_amdgcn_s_wait_asynccnt)
#define WAIT_ASYNC(n) __builtin_amdgcn_s_wait_asynccnt(n)
#else
#define WAIT_ASYNC(n) asm volatile("s_wait_asynccnt %0" ::"i"(n) : "memory")
#endif

// ---------------------------------------------------------------- fp32 -> bf16
__global__ void cvt_kernel(const float* __restrict__ x, __bf16* __restrict__ y,
                           int n4) {
  int i = blockIdx.x * blockDim.x + threadIdx.x;
  if (i < n4) {
    f32x4 v = ((const f32x4*)x)[i];
    bf16x4 o;
#pragma unroll
    for (int k = 0; k < 4; ++k) o[k] = (__bf16)v[k];
    ((bf16x4*)y)[i] = o;
  }
}

// ----------------------------------------------------- swapped-orientation GEMM
// D(1024 feat, ntok) = W(1024,1024) x X(ntok,1024)^T ; store D^T row-major
// (token, feature) with packed 16B/32B stores.  C = (x @ W.T + bias) * scale.
// Block 256 thr (8 waves), block tile 256(f) x 128(tok); wave tile 64x64.
enum { PK_BF16 = 0, PK_F32 = 1 };

template <int MODE>
__launch_bounds__(256)
__global__ void gemm_wt_kernel(const __bf16* __restrict__ W,
                               const __bf16* __restrict__ X,
                               const float* __restrict__ bias,  // per feature
                               void* __restrict__ Cout,         // (ntok,1024)
                               int nb, float scale) {
  const int lane = threadIdx.x & 31;
  const int wave = threadIdx.x >> 5;
  const int ln = lane & 15, hi = lane >> 4;
  const int bn = blockIdx.x % nb;
  const int bm = blockIdx.x / nb;
  const int m0 = bm * 256 + (wave >> 1) * 64;  // feature
  const int n0 = bn * 128 + (wave & 1) * 64;   // token

  v8f acc[4][4];
#pragma unroll
  for (int i = 0; i < 4; ++i)
#pragma unroll
    for (int j = 0; j < 4; ++j) acc[i][j] = vzero8();

  const __bf16* Ap = W + (size_t)(m0 + ln) * D + hi * 16;
  const __bf16* Bp = X + (size_t)(n0 + ln) * D + hi * 16;

  for (int k = 0; k < D; k += 32) {
    v16bf af[4];
#pragma unroll
    for (int i = 0; i < 4; ++i) af[i] = ld_frag_a(Ap + (size_t)(i * 16) * D + k);
#pragma unroll
    for (int j = 0; j < 4; ++j) {
      v16bf bfrag = *(const v16bf*)(Bp + (size_t)(j * 16) * D + k);
#pragma unroll
      for (int i = 0; i < 4; ++i) acc[i][j] = wmma_bf16(af[i], bfrag, acc[i][j]);
    }
  }

#pragma unroll
  for (int i = 0; i < 4; ++i) {
    const int f0 = m0 + 16 * i + 8 * hi;
    const f32x4 b0 = *(const f32x4*)(bias + f0);
    const f32x4 b1 = *(const f32x4*)(bias + f0 + 4);
#pragma unroll
    for (int j = 0; j < 4; ++j) {
      const int tok = n0 + 16 * j + ln;
      if constexpr (MODE == PK_BF16) {
        bf16x8 pk;
#pragma unroll
        for (int r = 0; r < 4; ++r) {
          pk[r]     = (__bf16)((acc[i][j][r]     + b0[r]) * scale);
          pk[4 + r] = (__bf16)((acc[i][j][4 + r] + b1[r]) * scale);
        }
        *(bf16x8*)((__bf16*)Cout + (size_t)tok * D + f0) = pk;
      } else {
        f32x4 o0, o1;
#pragma unroll
        for (int r = 0; r < 4; ++r) {
          o0[r] = (acc[i][j][r]     + b0[r]) * scale;
          o1[r] = (acc[i][j][4 + r] + b1[r]) * scale;
        }
        float* C = (float*)Cout + (size_t)tok * D + f0;
        *(f32x4*)C = o0;
        *(f32x4*)(C + 4) = o1;
      }
    }
  }
}

// ------------------------------------------------- v projection -> per-head v^T
// C = Xkv x Wv^T ; C-fragment stored transposed -> vT[b][h][d][s] (16B packed).
__launch_bounds__(256)
__global__ void gemm_vt_kernel(const __bf16* __restrict__ A,
                               const __bf16* __restrict__ Bt,
                               const float* __restrict__ bias,
                               __bf16* __restrict__ VT) {
  const int lane = threadIdx.x & 31;
  const int wave = threadIdx.x >> 5;
  const int ln = lane & 15, hi = lane >> 4;
  const int bn = blockIdx.x & 7;  // 8 N-blocks of 128 (N = 1024 features)
  const int bm = blockIdx.x >> 3;
  const int m0 = bm * 256 + (wave >> 1) * 64;  // token (within one batch)
  const int n0 = bn * 128 + (wave & 1) * 64;   // feature

  v8f acc[4][4];
#pragma unroll
  for (int i = 0; i < 4; ++i)
#pragma unroll
    for (int j = 0; j < 4; ++j) acc[i][j] = vzero8();

  const __bf16* Ap = A + (size_t)(m0 + ln) * D + hi * 16;
  const __bf16* Bp = Bt + (size_t)(n0 + ln) * D + hi * 16;

  for (int k = 0; k < D; k += 32) {
    v16bf af[4];
#pragma unroll
    for (int i = 0; i < 4; ++i) af[i] = ld_frag_a(Ap + (size_t)(i * 16) * D + k);
#pragma unroll
    for (int j = 0; j < 4; ++j) {
      v16bf bfrag = *(const v16bf*)(Bp + (size_t)(j * 16) * D + k);
#pragma unroll
      for (int i = 0; i < 4; ++i) acc[i][j] = wmma_bf16(af[i], bfrag, acc[i][j]);
    }
  }

  const int bidx = m0 / S;
#pragma unroll
  for (int i = 0; i < 4; ++i) {
#pragma unroll
    for (int j = 0; j < 4; ++j) {
      const int ng = n0 + 16 * j + ln;
      const int s = (m0 - bidx * S) + 16 * i + 8 * hi;  // 8 consecutive s
      const int hh = ng >> 6, dd = ng & 63;
      const float bb = bias[ng];
      bf16x8 pk;
#pragma unroll
      for (int r = 0; r < 8; ++r) pk[r] = (__bf16)(acc[i][j][r] + bb);
      *(bf16x8*)(VT + (size_t)((bidx * H + hh) * HD + dd) * S + s) = pk;
    }
  }
}

// ------------------------------------------------------- target mutual weights
__launch_bounds__(256)
__global__ void weights_kernel(const __bf16* __restrict__ TQ,
                               const __bf16* __restrict__ TK,
                               const float* __restrict__ mask,
                               float* __restrict__ Wout) {
  const int lane = threadIdx.x & 31;
  const int wave = threadIdx.x >> 5;
  const int ln = lane & 15, hi = lane >> 4;
  int idx = blockIdx.x;
  const int sb = idx & 7;  idx >>= 3;
  const int h  = idx & 15; idx >>= 4;
  const int b  = idx;
  const int s0 = sb * 128 + wave * 16;

  const __bf16* ap = TQ + (size_t)(b * S + s0 + ln) * D + h * HD + hi * 16;
  v16bf a0 = ld_frag_a(ap);
  v16bf a1 = ld_frag_a(ap + 32);

  v8f acc[4];
#pragma unroll
  for (int j = 0; j < 4; ++j) acc[j] = vzero8();
#pragma unroll
  for (int j = 0; j < 4; ++j) {
    const __bf16* kp = TK + (size_t)(b * TL + 16 * j + ln) * D + h * HD + hi * 16;
    acc[j] = wmma_bf16(a0, *(const v16bf*)kp, acc[j]);
    acc[j] = wmma_bf16(a1, *(const v16bf*)(kp + 32), acc[j]);
  }

  float wsum[8], msum[8];
#pragma unroll
  for (int r = 0; r < 8; ++r) { wsum[r] = 0.f; msum[r] = 0.f; }
#pragma unroll
  for (int j = 0; j < 4; ++j) {
    const int t = 16 * j + ln;
#pragma unroll
    for (int r = 0; r < 8; ++r) {
      const int s = s0 + r + 8 * hi;
      const float mv = mask[((size_t)b * S + s) * TL + t];
      wsum[r] += acc[j][r] * mv;
      msum[r] += mv;
    }
  }
#pragma unroll
  for (int r = 0; r < 8; ++r) {
    float w = wsum[r], m = msum[r];
    w += __shfl_xor(w, 1, 32); m += __shfl_xor(m, 1, 32);
    w += __shfl_xor(w, 2, 32); m += __shfl_xor(m, 2, 32);
    w += __shfl_xor(w, 4, 32); m += __shfl_xor(m, 4, 32);
    w += __shfl_xor(w, 8, 32); m += __shfl_xor(m, 8, 32);
    if (ln == 0)
      Wout[(size_t)(b * H + h) * S + s0 + r + 8 * hi] = w / m;
  }
}

// ------------------------------------------------------- fused flash attention
// K/V tiles are shared by all 8 waves of the block -> double-buffered async
// global->LDS staging (ASYNCcnt) with workgroup-barrier handshake.
__launch_bounds__(256)
__global__ void attn_kernel(const __bf16* __restrict__ Q,
                            const __bf16* __restrict__ Kd,
                            const __bf16* __restrict__ VT,
                            const float* __restrict__ Wgt,
                            __bf16* __restrict__ Out) {
  constexpr int KROW = 72;  // 64 data + pad -> 144B stride (conflict-free)
  constexpr int VROW = 40;  // 32 data + pad -> 80B stride  (conflict-free)
  __shared__ __bf16 Kt[2][32 * KROW];
  __shared__ __bf16 Vt[2][64 * VROW];
  __shared__ float  Pb[8][16 * 36];

  const int tid = threadIdx.x;
  const int lane = tid & 31;
  const int wave = tid >> 5;
  const int ln = lane & 15, hi = lane >> 4;
  int idx = blockIdx.x;
  const int tb = idx & 3;  idx >>= 2;
  const int h  = idx & 15; idx >>= 4;
  const int b  = idx;
  const int t0 = tb * 128 + wave * 16;

  const __bf16* Kbase = Kd + (size_t)(b * S) * D + h * HD;
  const __bf16* Vbase = VT + (size_t)((b * H + h) * HD) * S;
  const float*  Wb    = Wgt + (size_t)(b * H + h) * S;

  // cooperative staging addresses: 256 thr x 16B = one 4KB chunk per issue
  const int krow = tid >> 3, kch = tid & 7;  // K chunk: 32 rows x 64 bf16
  const int vrow = tid >> 2, vch = tid & 3;  // V chunk: 64 rows x 32 bf16
  const unsigned int kl0 = (unsigned int)(uintptr_t)(&Kt[0][0]) +
                           (unsigned int)(krow * KROW * 2 + kch * 16);
  const unsigned int vl0 = (unsigned int)(uintptr_t)(&Vt[0][0]) +
                           (unsigned int)(vrow * VROW * 2 + vch * 16);
  const unsigned int kbuf = 32 * KROW * 2, vbuf = 64 * VROW * 2;
  const __bf16* kg = Kbase + (size_t)krow * D + kch * 8;
  const __bf16* vg = Vbase + (size_t)vrow * S + vch * 8;

  auto issue = [&](int stage) {
    const int bsel = stage & 1;
    async_ld16(kl0 + bsel * kbuf, kg + (size_t)(stage * 32) * D);
    async_ld16(vl0 + bsel * vbuf, vg + stage * 32);
  };
  issue(0);
  issue(1);

  float* P = Pb[wave];
  const __bf16* qp = Q + (size_t)(b * T + t0 + ln) * D + h * HD + hi * 16;
  v16bf qa0 = ld_frag_a(qp);       // hd 0..31
  v16bf qa1 = ld_frag_a(qp + 32);  // hd 32..63

  v8f o[4];
#pragma unroll
  for (int dt = 0; dt < 4; ++dt) o[dt] = vzero8();
  float rmax[8], rsum[8];
#pragma unroll
  for (int r = 0; r < 8; ++r) { rmax[r] = -3.0e38f; rsum[r] = 0.f; }

  for (int it = 0; it < S / 32; ++it) {
    const int s0 = it * 32;
    const int bsel = it & 1;
    WAIT_ASYNC(2);       // my stage-it chunks landed
    __syncthreads();     // everyone's stage-it chunks landed

    v8f sc[2];
    sc[0] = vzero8(); sc[1] = vzero8();
#pragma unroll
    for (int j = 0; j < 2; ++j) {
      const __bf16* kr = &Kt[bsel][(16 * j + ln) * KROW + hi * 16];
      sc[j] = wmma_bf16(qa0, *(const v16bf*)kr, sc[j]);
      sc[j] = wmma_bf16(qa1, *(const v16bf*)(kr + 32), sc[j]);
      const float w = Wb[s0 + 16 * j + ln];
#pragma unroll
      for (int r = 0; r < 8; ++r) sc[j][r] *= w;
    }
#pragma unroll
    for (int r = 0; r < 8; ++r) {
      float tm = fmaxf(sc[0][r], sc[1][r]);
      tm = fmaxf(tm, __shfl_xor(tm, 1, 32));
      tm = fmaxf(tm, __shfl_xor(tm, 2, 32));
      tm = fmaxf(tm, __shfl_xor(tm, 4, 32));
      tm = fmaxf(tm, __shfl_xor(tm, 8, 32));
      const float nm = fmaxf(rmax[r], tm);
      const float alpha = __expf(rmax[r] - nm);
      const float p0 = __expf(sc[0][r] - nm);
      const float p1 = __expf(sc[1][r] - nm);
      float ts = p0 + p1;
      ts += __shfl_xor(ts, 1, 32);
      ts += __shfl_xor(ts, 2, 32);
      ts += __shfl_xor(ts, 4, 32);
      ts += __shfl_xor(ts, 8, 32);
      rsum[r] = rsum[r] * alpha + ts;
      rmax[r] = nm;
#pragma unroll
      for (int dt = 0; dt < 4; ++dt) o[dt][r] *= alpha;
      P[(r + 8 * hi) * 36 + ln] = p0;
      P[(r + 8 * hi) * 36 + 16 + ln] = p1;
    }
    // re-shape P (C-frag layout) -> A-frag layout via wave-private LDS
    const float* pr = P + ln * 36 + 16 * hi;
    f32x4 c0 = *(const f32x4*)(pr);
    f32x4 c1 = *(const f32x4*)(pr + 4);
    f32x4 c2 = *(const f32x4*)(pr + 8);
    f32x4 c3 = *(const f32x4*)(pr + 12);
    v16bf pa;
#pragma unroll
    for (int i = 0; i < 4; ++i) {
      pa[i] = (__bf16)c0[i];      pa[4 + i] = (__bf16)c1[i];
      pa[8 + i] = (__bf16)c2[i];  pa[12 + i] = (__bf16)c3[i];
    }
#pragma unroll
    for (int dt = 0; dt < 4; ++dt) {
      const __bf16* vr = &Vt[bsel][(16 * dt + ln) * VROW + hi * 16];
      o[dt] = wmma_bf16(pa, *(const v16bf*)vr, o[dt]);
    }
    __syncthreads();  // everyone done with buffer `bsel` before it is refilled
    if (it + 2 < S / 32) issue(it + 2);
  }

#pragma unroll
  for (int r = 0; r < 8; ++r) {
    const float inv = 1.0f / rsum[r];
    const int row = b * T + t0 + 8 * hi + r;
#pragma unroll
    for (int dt = 0; dt < 4; ++dt)
      Out[(size_t)row * D + h * HD + 16 * dt + ln] = (__bf16)(o[dt][r] * inv);
  }
}

// ------------------------------------------------------------------- dispatch
extern "C" void kernel_launch(void* const* d_in, const int* in_sizes, int n_in,
                              void* d_out, int out_size, void* d_ws, size_t ws_size,
                              hipStream_t stream) {
  (void)in_sizes; (void)n_in; (void)out_size; (void)ws_size;
  const float* hidden = (const float*)d_in[0];
  const float* kv     = (const float*)d_in[1];
  const float* targ   = (const float*)d_in[2];
  const float* mask   = (const float*)d_in[3];
  const float* Wq  = (const float*)d_in[4];  const float* bq  = (const float*)d_in[5];
  const float* Wk  = (const float*)d_in[6];  const float* bk  = (const float*)d_in[7];
  const float* Wv  = (const float*)d_in[8];  const float* bv  = (const float*)d_in[9];
  const float* Wwq = (const float*)d_in[10]; const float* bwq = (const float*)d_in[11];
  const float* Wwk = (const float*)d_in[12]; const float* bwk = (const float*)d_in[13];
  const float* Wo  = (const float*)d_in[14]; const float* bo  = (const float*)d_in[15];

  char* ws = (char*)d_ws;
  size_t off = 0;
  auto alloc = [&](size_t bytes) {
    void* p = ws + off;
    off += (bytes + 255) & ~(size_t)255;
    return p;
  };
  __bf16* Xh  = (__bf16*)alloc((size_t)BB * T * D * 2);
  __bf16* Xkv = (__bf16*)alloc((size_t)BB * S * D * 2);
  __bf16* Xt  = (__bf16*)alloc((size_t)BB * TL * D * 2);
  __bf16* Wqb  = (__bf16*)alloc((size_t)D * D * 2);
  __bf16* Wkb  = (__bf16*)alloc((size_t)D * D * 2);
  __bf16* Wvb  = (__bf16*)alloc((size_t)D * D * 2);
  __bf16* Wwqb = (__bf16*)alloc((size_t)D * D * 2);
  __bf16* Wwkb = (__bf16*)alloc((size_t)D * D * 2);
  __bf16* Wob  = (__bf16*)alloc((size_t)D * D * 2);
  __bf16* Qb  = (__bf16*)alloc((size_t)BB * T * D * 2);
  __bf16* Kb  = (__bf16*)alloc((size_t)BB * S * D * 2);
  __bf16* TQb = (__bf16*)alloc((size_t)BB * S * D * 2);
  __bf16* TKb = (__bf16*)alloc((size_t)BB * TL * D * 2);
  __bf16* VTb = (__bf16*)alloc((size_t)BB * S * D * 2);  // (B,H,64,S)
  float*  Wgt = (float*)alloc((size_t)BB * H * S * 4);
  __bf16* AOb = (__bf16*)alloc((size_t)BB * T * D * 2);

  auto cvt = [&](const float* src, __bf16* dst, int n) {
    int n4 = n / 4;
    cvt_kernel<<<(n4 + 255) / 256, 256, 0, stream>>>(src, dst, n4);
  };
  cvt(hidden, Xh, BB * T * D);
  cvt(kv, Xkv, BB * S * D);
  cvt(targ, Xt, BB * TL * D);
  cvt(Wq, Wqb, D * D);
  cvt(Wk, Wkb, D * D);
  cvt(Wv, Wvb, D * D);
  cvt(Wwq, Wwqb, D * D);
  cvt(Wwk, Wwkb, D * D);
  cvt(Wo, Wob, D * D);

  // projections: swapped-orientation NT GEMMs with packed transposed stores
  gemm_wt_kernel<PK_BF16><<<4 * (BB * T / 128), 256, 0, stream>>>(
      Wqb, Xh, bq, Qb, BB * T / 128, QSCALE);
  gemm_wt_kernel<PK_BF16><<<4 * (BB * S / 128), 256, 0, stream>>>(
      Wkb, Xkv, bk, Kb, BB * S / 128, 1.0f);
  gemm_wt_kernel<PK_BF16><<<4 * (BB * S / 128), 256, 0, stream>>>(
      Wwqb, Xkv, bwq, TQb, BB * S / 128, QSCALE);
  gemm_wt_kernel<PK_BF16><<<4 * (BB * TL / 128), 256, 0, stream>>>(
      Wwkb, Xt, bwk, TKb, BB * TL / 128, 1.0f);
  // v projection with per-head transposed output
  gemm_vt_kernel<<<(BB * S / 256) * 8, 256, 0, stream>>>(Xkv, Wvb, bv, VTb);

  // per-key weights from target mutual attention
  weights_kernel<<<BB * H * (S / 128), 256, 0, stream>>>(TQb, TKb, mask, Wgt);

  // fused flash attention (async-LDS staged K/V)
  attn_kernel<<<BB * H * (T / 128), 256, 0, stream>>>(Qb, Kb, VTb, Wgt, AOb);

  // output projection, packed fp32 result
  gemm_wt_kernel<PK_F32><<<4 * (BB * T / 128), 256, 0, stream>>>(
      Wob, AOb, bo, d_out, BB * T / 128, 1.0f);
}
